// GenEdge_64699387347548
// MI455X (gfx1250) — matrix operand order repaired and checked by reading, not compiled, not run on hardware
//
#include <hip/hip_runtime.h>
#include <hip/hip_bf16.h>
#include <hip/hip_fp16.h>

typedef _Float16 v16h __attribute__((ext_vector_type(16)));
typedef _Float16 v8h  __attribute__((ext_vector_type(8)));
typedef _Float16 v4h  __attribute__((ext_vector_type(4)));
typedef float    v8f  __attribute__((ext_vector_type(8)));

#define HH 400
#define WW 400
#define CC 128
#define PP 12

// ---------------------------------------------------------------- zero init
__global__ void k_zero(float* __restrict__ edge, unsigned* __restrict__ mx) {
    int i = blockIdx.x * blockDim.x + threadIdx.x;
    if (i < HH * WW) edge[i] = 0.0f;
    if (i == 0) *mx = 0u;
}

// ------------------------------------------- f32 -> f16 convert + L2 norms
__global__ __launch_bounds__(256) void k_prep(const float* __restrict__ cube,
                                              _Float16* __restrict__ hc,
                                              float* __restrict__ nrm) {
    const int lane = threadIdx.x & 31;
    const int pix  = blockIdx.x * 8 + (threadIdx.x >> 5);
    if (pix >= HH * WW) return;
    const float4 f = *(const float4*)(cube + (size_t)pix * CC + lane * 4);
    float ss = f.x * f.x + f.y * f.y + f.z * f.z + f.w * f.w;
#pragma unroll
    for (int m = 16; m >= 1; m >>= 1) ss += __shfl_xor(ss, m);
    if (lane == 0) nrm[pix] = sqrtf(ss);
    v4h hv = {(_Float16)f.x, (_Float16)f.y, (_Float16)f.z, (_Float16)f.w};
    *(v4h*)(hc + (size_t)pix * CC + lane * 4) = hv;
}

// ---------------------------------------------- fast arccos (raw v_sqrt)
__device__ __forceinline__ float acos_fast(float x) {
    float a = fabsf(x);
    float p = -0.0187293f;
    p = __builtin_fmaf(p, a, 0.0742610f);
    p = __builtin_fmaf(p, a, -0.2121144f);
    p = __builtin_fmaf(p, a, 1.5707288f);
    float r = __builtin_amdgcn_sqrtf(1.0f - a) * p;   // arg in [1e-6,2]: raw sqrt safe
    return x >= 0.0f ? r : 3.14159265358979f - r;
}

// -------- load one B tile (32x16 f16 per K-step, 4 K-steps) + neighbor norm
__device__ __forceinline__ void load_btile(const _Float16* __restrict__ hc,
                                           const float* __restrict__ nrm,
                                           int r, int col, size_t coff,
                                           v16h B[4], float& ns) {
    const _Float16* bb = hc + (size_t)r * WW * CC + coff;
    ns = nrm[(size_t)r * WW + col];
#pragma unroll
    for (int kc = 0; kc < 4; ++kc) {
        v8h lo = *(const v8h*)(bb + kc * 32);
        v8h hi = *(const v8h*)(bb + kc * 32 + 16);
        B[kc] = __builtin_shufflevector(lo, hi, 0, 1, 2, 3, 4, 5, 6, 7,
                                        8, 9, 10, 11, 12, 13, 14, 15);
    }
}

// -------- one di-step: 4 chained WMMAs (K=128) + packed SAD epilogue
__device__ __forceinline__ void sad_step(const v16h A[4], const v16h B[4],
                                         float ns, int di,
                                         const float ncm[8],
                                         const float kxc[8], const float kxs[8],
                                         const float madj[8], const float mbv[8],
                                         float gx[8], float gy[8]) {
    v8f acc = {};
#pragma unroll
    for (int kc = 0; kc < 4; ++kc)
        acc = __builtin_amdgcn_wmma_f32_16x16x32_f16(
            false, A[kc], false, B[kc], (short)0, acc, false, false);
    const int absdi = di < 0 ? -di : di;
    const int sdi   = (di > 0) - (di < 0);
    const float absdif = (float)absdi;
    const float s1f    = (float)(di + PP * sdi);
    const float sdif   = (float)sdi;
#pragma unroll
    for (int v = 0; v < 8; ++v) {
        float dn    = __builtin_fmaf(ncm[v], ns, 1e-4f);
        float ratio = acc[v] * __builtin_amdgcn_rcpf(dn);
        ratio = fminf(fmaxf(ratio, -0.999999f), 0.999999f);
        float sad = acos_fast(ratio);
        float kxw = __builtin_fmaf(-kxs[v], absdif, kxc[v]);
        float kyw = __builtin_fmaf(mbv[v], s1f, -sdif * madj[v]);
        gx[v] = __builtin_fmaf(sad, kxw, gx[v]);
        gy[v] = __builtin_fmaf(sad, kyw, gy[v]);
    }
}

// ------------------------------------------------- main WMMA SAD edge pass
// Wave = one output row h, 16 consecutive pixels [w0, w0+16).
// Band correlation: D[m,n] = A[w0+m]·S[w1+n] gives diagonals dj=(w1-w0)+n-m;
// 3 N-tiles (w1 = w0-12, w0+4, w0+20) cover dj in [-12,12].
// t OUTER (weight constants hoisted); di INNER, double-buffered B (ping/pong).
__global__ __launch_bounds__(128) void k_sad(const _Float16* __restrict__ hc,
                                             const float* __restrict__ nrm,
                                             float* __restrict__ edge) {
    const int lane = threadIdx.x & 31;
    const int g    = lane >> 4;        // half-wave group
    const int nx   = lane & 15;        // N index (and A row index)
    const int pd   = nx - 8 * g;       // per-lane part of dj
    const int h    = PP + blockIdx.y * 4 + (threadIdx.x >> 5);
    const int w0   = PP + blockIdx.x * 16;

    // A tile: 16x128 f16, WMMA A layout (lane group g selects K sub-blocks)
    v16h A[4];
    {
        const _Float16* base = hc + ((size_t)h * WW + (w0 + nx)) * CC;
#pragma unroll
        for (int kc = 0; kc < 4; ++kc) {
            const _Float16* p0 = base + kc * 32 + 8 * g;
            v8h lo = *(const v8h*)(p0);
            v8h hi = *(const v8h*)(p0 + 16);
            A[kc] = __builtin_shufflevector(lo, hi, 0, 1, 2, 3, 4, 5, 6, 7,
                                            8, 9, 10, 11, 12, 13, 14, 15);
        }
    }
    float ncm[8];
#pragma unroll
    for (int v = 0; v < 8; ++v) ncm[v] = nrm[(size_t)h * WW + w0 + v + 8 * g];

    float gx[8], gy[8];
#pragma unroll
    for (int v = 0; v < 8; ++v) { gx[v] = 0.0f; gy[v] = 0.0f; }

#pragma unroll
    for (int t = 0; t < 3; ++t) {
        const int w1 = w0 - PP + 16 * t;
        int col = w1 + nx;
        col = col > (WW - 1) ? (WW - 1) : col;  // clamp; tainted pairs masked at store
        const size_t coff = (size_t)col * CC + 8 * g;

        // hoisted per-(lane,v) weight constants (invariant over di):
        //   kx = mb*(dj+12*sdj) - mb*sdj*|di| ;  ky = mb*(di+12*sdi) - sdi*mb*|dj|
        float kxc[8], kxs[8], madj[8], mbv[8];
#pragma unroll
        for (int v = 0; v < 8; ++v) {
            const int dj  = 16 * t - PP + pd - v;
            const int adj = dj < 0 ? -dj : dj;
            const int sdj = (dj > 0) - (dj < 0);
            const float mb = (adj <= PP) ? 1.0f : 0.0f;
            kxc[v]  = mb * (float)(dj + PP * sdj);
            kxs[v]  = mb * (float)sdj;
            madj[v] = mb * (float)adj;
            mbv[v]  = mb;
        }

        // software pipeline: while WMMAs+epilogue run on one buffer,
        // the next neighbor row's 8x b128 loads are in flight.
        v16h Bc[4], Bn[4];
        float nsc, nsn;
        load_btile(hc, nrm, h - PP, col, coff, Bc, nsc);
#pragma unroll 1
        for (int dd = 0; dd < 12; ++dd) {
            const int di0 = -PP + 2 * dd;
            load_btile(hc, nrm, h + di0 + 1, col, coff, Bn, nsn);
            sad_step(A, Bc, nsc, di0, ncm, kxc, kxs, madj, mbv, gx, gy);
            load_btile(hc, nrm, h + di0 + 2, col, coff, Bc, nsc);
            sad_step(A, Bn, nsn, di0 + 1, ncm, kxc, kxs, madj, mbv, gx, gy);
        }
        sad_step(A, Bc, nsc, PP, ncm, kxc, kxs, madj, mbv, gx, gy);  // di = +12
    }

    // reduce across the 16 N-lanes, store |gx|+|gy| for interior pixels
#pragma unroll
    for (int v = 0; v < 8; ++v) {
        float sx = gx[v], sy = gy[v];
#pragma unroll
        for (int m = 8; m >= 1; m >>= 1) {
            sx += __shfl_xor(sx, m);
            sy += __shfl_xor(sy, m);
        }
        const int w = w0 + v + 8 * g;
        if (nx == 0 && w < WW - PP)
            edge[(size_t)h * WW + w] = fabsf(sx) + fabsf(sy);
    }
}

// ------------------------------------------------------------- global max
__global__ void k_max(const float* __restrict__ edge, unsigned* __restrict__ mx) {
    float m = 0.0f;
    for (int i = blockIdx.x * blockDim.x + threadIdx.x; i < HH * WW;
         i += gridDim.x * blockDim.x)
        m = fmaxf(m, edge[i]);
#pragma unroll
    for (int s = 16; s >= 1; s >>= 1) m = fmaxf(m, __shfl_xor(m, s));
    if ((threadIdx.x & 31) == 0) atomicMax(mx, __float_as_uint(m));  // all vals >= 0
}

// -------------------------------------------------------------- normalize
__global__ void k_norm(const float* __restrict__ edge,
                       const unsigned* __restrict__ mx,
                       float* __restrict__ out) {
    int i = blockIdx.x * blockDim.x + threadIdx.x;
    if (i >= HH * WW) return;
    out[i] = edge[i] * (1.0f / __uint_as_float(*mx));
}

extern "C" void kernel_launch(void* const* d_in, const int* in_sizes, int n_in,
                              void* d_out, int out_size, void* d_ws, size_t ws_size,
                              hipStream_t stream) {
    const float* cube = (const float*)d_in[0];
    float* out = (float*)d_out;

    char* ws = (char*)d_ws;
    size_t off = 0;
    _Float16* hc = (_Float16*)(ws + off);
    off += (size_t)HH * WW * CC * sizeof(_Float16);      // 40.96 MB f16 cube
    off = (off + 255) & ~(size_t)255;
    float* nrm = (float*)(ws + off);
    off += (size_t)HH * WW * sizeof(float);              // per-pixel norms
    off = (off + 255) & ~(size_t)255;
    float* edge = (float*)(ws + off);
    off += (size_t)HH * WW * sizeof(float);              // unnormalized edge map
    off = (off + 255) & ~(size_t)255;
    unsigned* mx = (unsigned*)(ws + off);

    k_zero<<<(HH * WW + 255) / 256, 256, 0, stream>>>(edge, mx);
    k_prep<<<(HH * WW + 7) / 8, 256, 0, stream>>>(cube, hc, nrm);

    dim3 grid(((WW - 2 * PP) + 15) / 16, (HH - 2 * PP) / 4);  // 24 x 94
    k_sad<<<grid, 128, 0, stream>>>(hc, nrm, edge);

    k_max<<<64, 256, 0, stream>>>(edge, mx);
    k_norm<<<(HH * WW + 255) / 256, 256, 0, stream>>>(edge, mx, out);
}
